// WeightedAttention_5600637354348
// MI455X (gfx1250) — compile-verified, matrix operand order checked by reading, not run on hardware
//
#include <hip/hip_runtime.h>
#include <hip/hip_bf16.h>

typedef __bf16 bf16;
typedef __attribute__((ext_vector_type(16))) __bf16 v16bf;
typedef __attribute__((ext_vector_type(8)))  float  v8f;

// ---------------------------------------------------------------------------
// WMMA helpers (CDNA5 wave32, v_wmma_f32_16x16x32_bf16)
//   A (16x32): lane L = row M=L%16; half i holds K = (i/8)*16 + (L/16)*8 + i%8
//   B (32x16): lane L = col N=L%16; same K pattern.
//   C/D (16x16 f32): lane L = col N=L%16; vgpr r = row M = r + (L/16)*8.
// ---------------------------------------------------------------------------

static __device__ inline v8f wmma_bf16(v16bf a, v16bf b, v8f c) {
  return __builtin_amdgcn_wmma_f32_16x16x32_bf16(
      /*neg_a=*/false, a, /*neg_b=*/false, b,
      /*c_mod=*/(short)0, c, /*reuse_a=*/false, /*reuse_b=*/false);
}

// Fragment from row-major bf16 where K runs along the row (contiguous 8-elt
// runs -> two 16B loads). Works as A-fragment (rows = M) or as B-fragment for
// a transposed operand (rows = N).
static __device__ inline v16bf frag_rowmajor_bf16(const bf16* p, int ld, int m0, int k0) {
  const int lane = threadIdx.x & 31;
  const bf16* row = p + (size_t)(m0 + (lane & 15)) * ld + k0 + ((lane >> 4) << 3);
  v16bf f;
#pragma unroll
  for (int i = 0; i < 8; ++i) {
    f[i]     = row[i];
    f[i + 8] = row[16 + i];
  }
  return f;
}

// ---------------------------------------------------------------------------
// One-time layout/precision fixups (streamed once; negligible vs GEMM time).
// ---------------------------------------------------------------------------

// fp32 -> bf16, 4 elements per thread.
__global__ void cvt_bf16_kernel(const float* __restrict__ src,
                                bf16* __restrict__ dst, int n4) {
  int i = blockIdx.x * blockDim.x + threadIdx.x;
  if (i < n4) {
    const float4 v = ((const float4*)src)[i];
    bf16* o = dst + (size_t)i * 4;
    o[0] = (bf16)v.x; o[1] = (bf16)v.y; o[2] = (bf16)v.z; o[3] = (bf16)v.w;
  }
}

// Wf f32 [H=16][64][1024] -> WfT bf16 [H][1024][64]
__global__ void wf_transpose_kernel(const float* __restrict__ wf,
                                    bf16* __restrict__ wfT) {
  int i = blockIdx.x * blockDim.x + threadIdx.x;   // over 16*1024*64 outputs
  const int d = i & 63;
  const int e = (i >> 6) & 1023;
  const int h = i >> 16;
  wfT[i] = (bf16)wf[((size_t)h * 64 + d) * 1024 + e];
}

// ---------------------------------------------------------------------------
// Kernel: P = X @ W^T + bias (all bf16 operands, f32 accum), store bf16.
// X: [4096 x 1024] bf16, W: [1024 x 1024] bf16.
// Block = 512 thr (16 waves); wave w -> N strip [w*64, w*64+64),
// blockIdx.x -> M tile of 16 rows. grid = 256.
// vtranspose==0 -> dst [B,H,S,64];  vtranspose==1 -> dst [B,H,64,S].
// ---------------------------------------------------------------------------
__global__ void proj_bf16_kernel(const bf16* __restrict__ X,
                                 const bf16* __restrict__ W,
                                 const float* __restrict__ bias,
                                 bf16* __restrict__ dst,
                                 float scale, int vtranspose) {
  const int wave = threadIdx.x >> 5;
  const int lane = threadIdx.x & 31;
  const int m0 = blockIdx.x * 16;
  const int n0 = wave * 64;

  v8f acc[4] = {};
  for (int k0 = 0; k0 < 1024; k0 += 32) {
    v16bf a = frag_rowmajor_bf16(X, 1024, m0, k0);
#pragma unroll
    for (int nt = 0; nt < 4; ++nt) {
      // B[k,n] = W[n,k]: transposed operand, W rows contiguous in k.
      v16bf bw = frag_rowmajor_bf16(W, 1024, n0 + nt * 16, k0);
      acc[nt] = wmma_bf16(a, bw, acc[nt]);
    }
  }

#pragma unroll
  for (int nt = 0; nt < 4; ++nt) {
    const int n = n0 + nt * 16 + (lane & 15);
    const float bn = bias[n];
    const int h = n >> 6;     // head
    const int d = n & 63;     // dim within head
#pragma unroll
    for (int r = 0; r < 8; ++r) {
      const int m = m0 + r + ((lane >> 4) << 3);
      const int bb = m >> 11;      // batch (S = 2048)
      const int s  = m & 2047;
      const bf16 v = (bf16)((acc[nt][r] + bn) * scale);
      if (vtranspose)
        dst[(((size_t)bb * 16 + h) * 64 + d) * 2048 + s] = v;
      else
        dst[(((size_t)bb * 16 + h) * 2048 + s) * 64 + d] = v;
    }
  }
}

// ---------------------------------------------------------------------------
// Attention: scores -> softmax over HEADS -> attn (global) -> context ->
// per-head final linear * softmax(kappa).
// Block = 512 thr (16 waves; wave h == head h). blockIdx = (q-tile, batch).
// Dynamic LDS 64 KB: scores/attn staging [16h][16q][32k] f32,
// then reused for context transpose [16h][16q][64d] f32.
// ---------------------------------------------------------------------------
__global__ void attn_softmaxH_kernel(const bf16* __restrict__ qb,
                                     const bf16* __restrict__ kbuf,
                                     const bf16* __restrict__ vT,     // [B,H,64,S]
                                     const unsigned char* __restrict__ mask,
                                     const float* __restrict__ kappa_w,
                                     const bf16* __restrict__ wfT,    // [H,1024,64]
                                     const float* __restrict__ bfb,
                                     float* __restrict__ out,
                                     float* __restrict__ attn_out) {
  extern __shared__ float lds[];   // 16384 floats = 64 KB
  const int h    = threadIdx.x >> 5;   // wave id == head
  const int lane = threadIdx.x & 31;
  const int b    = blockIdx.y;
  const int q0   = blockIdx.x * 16;

  const bf16* qh  = qb   + (((size_t)b * 16 + h) * 2048 + q0) * 64;
  const bf16* kh  = kbuf + ((size_t)b * 16 + h) * 2048 * 64;
  const bf16* vhT = vT   + ((size_t)b * 16 + h) * 64 * 2048;

  // q tile [16 x 64] resident for the whole k loop.
  const v16bf aq0 = frag_rowmajor_bf16(qh, 64, 0, 0);
  const v16bf aq1 = frag_rowmajor_bf16(qh, 64, 0, 32);

  v8f ctx[4] = {};

  for (int kc = 0; kc < 2048; kc += 32) {
    if (kc + 32 < 2048) {  // prefetch next streaming chunk (k rows + vT cols)
      __builtin_prefetch(kh + (size_t)(kc + 32) * 64, 0, 0);
      __builtin_prefetch(vhT + (size_t)(lane & 15) * 2048 + kc + 32, 0, 0);
    }
    // ---- scores: q[16x64] @ k[kc:kc+32]^T -> two 16x16 f32 tiles ----
    v8f sc0 = {}, sc1 = {};
    {
      v16bf bk0a = frag_rowmajor_bf16(kh, 64, kc, 0);
      v16bf bk0b = frag_rowmajor_bf16(kh, 64, kc, 32);
      sc0 = wmma_bf16(aq0, bk0a, sc0);
      sc0 = wmma_bf16(aq1, bk0b, sc0);
      v16bf bk1a = frag_rowmajor_bf16(kh, 64, kc + 16, 0);
      v16bf bk1b = frag_rowmajor_bf16(kh, 64, kc + 16, 32);
      sc1 = wmma_bf16(aq0, bk1a, sc1);
      sc1 = wmma_bf16(aq1, bk1b, sc1);
    }
    // stage this head's score tile [16q x 32k] into LDS region h
    {
      float* reg = lds + h * 512;
#pragma unroll
      for (int r = 0; r < 8; ++r) {
        const int m = r + ((lane >> 4) << 3);
        reg[m * 32 + (lane & 15)]      = sc0[r];
        reg[m * 32 + 16 + (lane & 15)] = sc1[r];
      }
    }
    __syncthreads();

    // ---- softmax across the 16 heads: thread owns one (qi, ki) point ----
    {
      const int qi = h;      // 16 waves x 32 lanes == 16q x 32k points
      const int ki = lane;
      const int base = qi * 32 + ki;
      const bool msk = mask[(size_t)b * 2048 + kc + ki] != 0;
      float x[16];
#pragma unroll
      for (int hh = 0; hh < 16; ++hh) {
        const float v = lds[hh * 512 + base];
        x[hh] = msk ? -1e18f : v;
      }
      float mx = x[0];
#pragma unroll
      for (int hh = 1; hh < 16; ++hh) mx = fmaxf(mx, x[hh]);
      float sum = 0.f;
#pragma unroll
      for (int hh = 0; hh < 16; ++hh) { x[hh] = __expf(x[hh] - mx); sum += x[hh]; }
      const float inv = 1.0f / sum;
#pragma unroll
      for (int hh = 0; hh < 16; ++hh) {
        const float a = x[hh] * inv;
        lds[hh * 512 + base] = a;
        attn_out[(((size_t)b * 16 + hh) * 2048 + q0 + qi) * 2048 + kc + ki] = a;
      }
    }
    __syncthreads();

    // ---- context += attn[16x32] @ v[kc:kc+32, 0:64] (vT rows = d) ----
    v16bf aat;
    {
      const float* ar = lds + h * 512 + (lane & 15) * 32 + ((lane >> 4) << 3);
#pragma unroll
      for (int i = 0; i < 8; ++i) {
        aat[i]     = (bf16)ar[i];
        aat[i + 8] = (bf16)ar[16 + i];
      }
    }
#pragma unroll
    for (int nt = 0; nt < 4; ++nt) {
      // B[k, n=d] = vT[d][kc+k]: transposed operand, rows contiguous in k.
      v16bf bv = frag_rowmajor_bf16(vhT, 2048, nt * 16, kc);
      ctx[nt] = wmma_bf16(aat, bv, ctx[nt]);
    }
  }

  // ---- transpose context (C-layout) -> A-layout via LDS ----
  __syncthreads();
  float* cst = lds + h * 1024;   // 16 rows x 64 cols per head
#pragma unroll
  for (int nt = 0; nt < 4; ++nt) {
#pragma unroll
    for (int r = 0; r < 8; ++r) {
      cst[(r + ((lane >> 4) << 3)) * 64 + nt * 16 + (lane & 15)] = ctx[nt][r];
    }
  }
  v16bf ac0, ac1;
  {
    const float* ar = cst + (lane & 15) * 64 + ((lane >> 4) << 3);
#pragma unroll
    for (int i = 0; i < 8; ++i) {
      ac0[i]     = (bf16)ar[i];        ac0[i + 8] = (bf16)ar[16 + i];
      ac1[i]     = (bf16)ar[32 + i];   ac1[i + 8] = (bf16)ar[48 + i];
    }
  }

  // ---- kappa softmax (16 scalars) ----
  float kmax = kappa_w[0];
#pragma unroll
  for (int i = 1; i < 16; ++i) kmax = fmaxf(kmax, kappa_w[i]);
  float ksum = 0.f, kex = 0.f;
#pragma unroll
  for (int i = 0; i < 16; ++i) {
    const float e = __expf(kappa_w[i] - kmax);
    ksum += e;
    if (i == h) kex = e;
  }
  const float kap = kex / ksum;

  // ---- out[b,h,q0:q0+16,:] = (context @ Wf[h] + bf[h]) * kappa[h] ----
  // B[k=d, n=e] = WfT[h][e][d]: transposed operand, WfT rows contiguous in d.
  const bf16* wfTh = wfT + (size_t)h * 1024 * 64;
  const float* bfh = bfb + (size_t)h * 1024;
  float* outh = out + (((size_t)b * 16 + h) * 2048 + q0) * 1024;
  for (int e0 = 0; e0 < 1024; e0 += 16) {
    v8f acc = {};
    acc = wmma_bf16(ac0, frag_rowmajor_bf16(wfTh, 64, e0, 0),  acc);
    acc = wmma_bf16(ac1, frag_rowmajor_bf16(wfTh, 64, e0, 32), acc);
    const float bn = bfh[e0 + (lane & 15)];
#pragma unroll
    for (int r = 0; r < 8; ++r) {
      const int m = r + ((lane >> 4) << 3);
      outh[(size_t)m * 1024 + e0 + (lane & 15)] = (acc[r] + bn) * kap;
    }
  }
}

// ---------------------------------------------------------------------------
extern "C" void kernel_launch(void* const* d_in, const int* in_sizes, int n_in,
                              void* d_out, int out_size, void* d_ws, size_t ws_size,
                              hipStream_t stream) {
  (void)in_sizes; (void)n_in; (void)out_size; (void)ws_size;

  const float* query = (const float*)d_in[0];
  const float* key_  = (const float*)d_in[1];
  const float* value = (const float*)d_in[2];
  const unsigned char* mask = (const unsigned char*)d_in[3];
  const float* Wq = (const float*)d_in[4];
  const float* bq = (const float*)d_in[5];
  const float* Wk = (const float*)d_in[6];
  const float* bk = (const float*)d_in[7];
  const float* Wv = (const float*)d_in[8];
  const float* bv = (const float*)d_in[9];
  const float* kappa_w = (const float*)d_in[10];
  const float* Wf  = (const float*)d_in[11];
  const float* bfb = (const float*)d_in[12];

  float* out  = (float*)d_out;                            // [2,16,2048,1024]
  float* attn = out + (size_t)2 * 16 * 2048 * 1024;       // [2,16,2048,2048]

  const size_t ACT = (size_t)2 * 2048 * 1024;             // 4 Mi elems
  const size_t WEL = (size_t)1024 * 1024;                 // 1 Mi elems
  bf16* qb  = (bf16*)d_ws;            // [B,H,S,64]
  bf16* kb  = qb  + ACT;              // [B,H,S,64]
  bf16* vTb = kb  + ACT;              // [B,H,64,S]
  bf16* xq  = vTb + ACT;              // bf16 copies of activations
  bf16* xk  = xq  + ACT;
  bf16* xv  = xk  + ACT;
  bf16* wqb = xv  + ACT;              // bf16 copies of weights
  bf16* wkb = wqb + WEL;
  bf16* wvb = wkb + WEL;
  bf16* wfT = wvb + WEL;              // [H,1024,64] bf16

  // One-time converts / transposes (streamed once, bandwidth-trivial).
  const int actBlocks = (int)(ACT / 4 / 256);
  const int wBlocks   = (int)(WEL / 4 / 256);
  cvt_bf16_kernel<<<actBlocks, 256, 0, stream>>>(query, xq, (int)(ACT / 4));
  cvt_bf16_kernel<<<actBlocks, 256, 0, stream>>>(key_,  xk, (int)(ACT / 4));
  cvt_bf16_kernel<<<actBlocks, 256, 0, stream>>>(value, xv, (int)(ACT / 4));
  cvt_bf16_kernel<<<wBlocks, 256, 0, stream>>>(Wq, wqb, (int)(WEL / 4));
  cvt_bf16_kernel<<<wBlocks, 256, 0, stream>>>(Wk, wkb, (int)(WEL / 4));
  cvt_bf16_kernel<<<wBlocks, 256, 0, stream>>>(Wv, wvb, (int)(WEL / 4));
  wf_transpose_kernel<<<(16 * 1024 * 64) / 256, 256, 0, stream>>>(Wf, wfT);

  // Projections: 256 blocks x 16 waves cover [4096 x 1024].
  // q gets 1/sqrt(dph) folded in; v is written transposed [B,H,64,S].
  proj_bf16_kernel<<<256, 512, 0, stream>>>(xq, wqb, bq, qb,  0.125f, 0);
  proj_bf16_kernel<<<256, 512, 0, stream>>>(xk, wkb, bk, kb,  1.0f,   0);
  proj_bf16_kernel<<<256, 512, 0, stream>>>(xv, wvb, bv, vTb, 1.0f,   1);

  // Attention: grid = (S/16 q-tiles, B), block = 16 waves (one per head),
  // 64 KB dynamic LDS for the cross-head softmax exchange.
  dim3 grid(128, 2);
  attn_softmaxH_kernel<<<grid, 512, 65536, stream>>>(qb, kb, vTb, mask, kappa_w,
                                                     wfT, bfb, out, attn);
}
